// HashingMemory_35390530519795
// MI455X (gfx1250) — compile-verified, matrix operand order checked by reading, not run on hardware
//
#include <hip/hip_runtime.h>
#include <math.h>

#define BS_      16384
#define IN_DIM_  1024
#define KDIM_    256
#define DEPTH_   2
#define NQ_      (DEPTH_*KDIM_)   // 512
#define NKEYS_   1000
#define TOPK_    32
#define EPS_     1e-10f
#define NEGINF_  (-3.0e38f)

typedef __attribute__((ext_vector_type(16))) __bf16 v16bf;
typedef __attribute__((ext_vector_type(8)))  __bf16 v8bf;
typedef __attribute__((ext_vector_type(4)))  __bf16 v4bf;
typedef __attribute__((ext_vector_type(2)))  __bf16 v2bf;
typedef __attribute__((ext_vector_type(8)))  float  v8f;

// Load a 16-element bf16 WMMA fragment: two contiguous 8-element (16B) chunks
// at p and p+16, per the ISA 16-bit A/B VGPR layout (K groups {g*8.., 16+g*8..}).
static __device__ __forceinline__ v16bf load_frag(const __bf16* p) {
  v8bf lo = *(const v8bf*)(p);
  v8bf hi = *(const v8bf*)(p + 16);
  return __builtin_shufflevector(lo, hi, 0,1,2,3,4,5,6,7,8,9,10,11,12,13,14,15);
}

// Async copy 16 bytes global -> LDS, tracked by ASYNCcnt (ISA 08_async_tensor §4).
static __device__ __forceinline__ void async_copy_b128(void* lds, const void* gptr) {
  unsigned l = (unsigned)(size_t)lds;                 // low 32 bits == LDS offset
  unsigned long long g = (unsigned long long)(size_t)gptr;
  asm volatile("global_load_async_to_lds_b128 %0, %1, off" :: "v"(l), "v"(g) : "memory");
}

// ---------- Kernel 0a: W f32 -> bf16 (one-time, reused by 512 blocks) ----------
__global__ void k_cvt_w(const float* __restrict__ W, __bf16* __restrict__ Wbf) {
  int idx = (blockIdx.x * 256 + threadIdx.x) * 4;   // 512 blocks * 256 * 4 = 512*1024
  float4 v = *(const float4*)(W + idx);
  v4bf o = { (__bf16)v.x, (__bf16)v.y, (__bf16)v.z, (__bf16)v.w };
  *(v4bf*)(Wbf + idx) = o;
}

// ---------- Kernel 0b: keys -> normalized bf16, layout [depth][key][k] ----------
__global__ void k_keys(const float* __restrict__ keys, __bf16* __restrict__ keysn) {
  int d = blockIdx.x / NKEYS_;
  int key = blockIdx.x % NKEYS_;
  int t = threadIdx.x;                               // 256 threads == KDIM_
  __shared__ float red[256];
  float v = keys[((size_t)key * DEPTH_ + d) * KDIM_ + t];
  red[t] = v * v;
  __syncthreads();
  for (int s = 128; s > 0; s >>= 1) {
    if (t < s) red[t] += red[t + s];
    __syncthreads();
  }
  float rn = 1.0f / (sqrtf(red[0]) + EPS_);          // folds 1/(||key||+eps) into key
  keysn[((size_t)d * NKEYS_ + key) * KDIM_ + t] = (__bf16)(v * rn);
}

// ---------- Kernel 1: q = relu(x @ W^T + b), per-depth L2-normalize -> qn bf16 ----------
// Block: 32 rows x 512 cols, 256 threads (8 waves); wave owns 2 M-tiles x 4 N-tiles.
// x staged via double-buffered async-to-LDS (64-wide K chunks, 2 b128 asyncs/thread).
// LDS ~80KB: legal on gfx1250 (320KB/WGP), 3 blocks resident per WGP.
__global__ void k_query(const float* __restrict__ x, const __bf16* __restrict__ Wbf,
                        const float* __restrict__ b, __bf16* __restrict__ qn) {
  __shared__ __align__(16) float As2[2][32][64];     // async-staged x (f32), 16KB
  __shared__ float qbuf[32][NQ_];                    // relu output, 64KB
  __shared__ float rnorm[64];                        // 32 rows x 2 depths
  const int t = threadIdx.x;
  const int wv = t >> 5, lane = t & 31;
  const int m = lane & 15, g = lane >> 4;
  const int rb = blockIdx.x * 32;
  const int sr = t >> 3, sc = (t & 7) * 8;           // staging map: 32B per thread

  v8f acc[2][4];
  #pragma unroll
  for (int mt = 0; mt < 2; ++mt)
    #pragma unroll
    for (int i = 0; i < 4; ++i) acc[mt][i] = (v8f){0,0,0,0,0,0,0,0};

  // prologue: fill both buffers (2 asyncs per thread per chunk)
  #pragma unroll
  for (int p = 0; p < 2; ++p) {
    async_copy_b128(&As2[p][sr][sc],     x + (size_t)(rb + sr) * IN_DIM_ + p * 64 + sc);
    async_copy_b128(&As2[p][sr][sc + 4], x + (size_t)(rb + sr) * IN_DIM_ + p * 64 + sc + 4);
  }

  for (int ch = 0; ch < 16; ++ch) {                  // 16 chunks of K=64
    if (ch < 15) asm volatile("s_wait_asynccnt 0x2" ::: "memory");
    else         asm volatile("s_wait_asynccnt 0x0" ::: "memory");
    __syncthreads();
    const int cur = ch & 1;
    #pragma unroll
    for (int ks = 0; ks < 2; ++ks) {                 // two 32-wide K steps per chunk
      // build bf16 A fragments (2 M-tiles) from staged f32
      v16bf af2[2];
      #pragma unroll
      for (int mt = 0; mt < 2; ++mt) {
        const float* ar = &As2[cur][mt * 16 + m][ks * 32 + g * 8];
        float4 f0 = *(const float4*)(ar);
        float4 f1 = *(const float4*)(ar + 4);
        float4 f2 = *(const float4*)(ar + 16);
        float4 f3 = *(const float4*)(ar + 20);
        v16bf af;
        af[0]=(__bf16)f0.x;  af[1]=(__bf16)f0.y;  af[2]=(__bf16)f0.z;  af[3]=(__bf16)f0.w;
        af[4]=(__bf16)f1.x;  af[5]=(__bf16)f1.y;  af[6]=(__bf16)f1.z;  af[7]=(__bf16)f1.w;
        af[8]=(__bf16)f2.x;  af[9]=(__bf16)f2.y;  af[10]=(__bf16)f2.z; af[11]=(__bf16)f2.w;
        af[12]=(__bf16)f3.x; af[13]=(__bf16)f3.y; af[14]=(__bf16)f3.z; af[15]=(__bf16)f3.w;
        af2[mt] = af;
      }
      // batch all 4 B fragments, each feeds both M-tiles (8 WMMAs)
      v16bf bq[4];
      #pragma unroll
      for (int i = 0; i < 4; ++i) {
        int n = (wv * 4 + i) * 16 + m;
        bq[i] = load_frag(Wbf + (size_t)n * IN_DIM_ + ch * 64 + ks * 32 + g * 8);
      }
      #pragma unroll
      for (int i = 0; i < 4; ++i)
        #pragma unroll
        for (int mt = 0; mt < 2; ++mt)
          acc[mt][i] = __builtin_amdgcn_wmma_f32_16x16x32_bf16(
              false, af2[mt], false, bq[i], (short)0, acc[mt][i], false, false);
    }
    __syncthreads();                                 // WAR: computes done before refill
    if (ch + 2 < 16) {
      async_copy_b128(&As2[cur][sr][sc],
                      x + (size_t)(rb + sr) * IN_DIM_ + (ch + 2) * 64 + sc);
      async_copy_b128(&As2[cur][sr][sc + 4],
                      x + (size_t)(rb + sr) * IN_DIM_ + (ch + 2) * 64 + sc + 4);
    }
  }
  // bias + relu; C layout: M = mt*16 + vr + 8*g, N = lane&15
  #pragma unroll
  for (int i = 0; i < 4; ++i) {
    int n = (wv * 4 + i) * 16 + m;
    float bias = b[n];
    #pragma unroll
    for (int mt = 0; mt < 2; ++mt)
      #pragma unroll
      for (int vr = 0; vr < 8; ++vr)
        qbuf[mt * 16 + vr + 8 * g][n] = fmaxf(acc[mt][i][vr] + bias, 0.0f);
  }
  __syncthreads();
  {  // per (row, depth) L2 norms: 4 lanes per pair, width-4 shuffle reduce
    int p = t >> 2, l4 = t & 3;
    int row = p >> 1, d = p & 1;
    float s = 0.f;
    for (int i = 0; i < 64; ++i) {
      float v = qbuf[row][d * KDIM_ + l4 + i * 4];
      s += v * v;
    }
    s += __shfl_xor(s, 1, 4);
    s += __shfl_xor(s, 2, 4);
    if (l4 == 0) rnorm[p] = 1.0f / (sqrtf(s) + EPS_);
  }
  __syncthreads();
  for (int i = 0; i < 32; ++i) {   // 32*512 elems, 2 per thread per iter (packed store)
    int idx = t + i * 256;
    int row = idx >> 8, col = (idx & 255) * 2;
    float rn = rnorm[row * 2 + (col >> 8)];
    v2bf o = { (__bf16)(qbuf[row][col] * rn), (__bf16)(qbuf[row][col + 1] * rn) };
    *(v2bf*)(qn + (size_t)(rb + row) * NQ_ + col) = o;
  }
}

// ---------- Kernel 2: logits GEMM + per-depth top32 + cartesian combine + softmax ----------
// Block: 16 batch rows (full WMMA M dimension), 256 threads (8 waves).
// Top-k phases: 16 lanes per row, width-16 shuffle argmax. LDS ~72KB (gfx1250 big-LDS).
__global__ void k_logits_topk(const __bf16* __restrict__ qn, const __bf16* __restrict__ keysn,
                              float* __restrict__ out) {
  __shared__ float logits[16][1024];                 // 64KB; reused as candidate buffer
  __shared__ float s0[16][TOPK_], s1[16][TOPK_];
  __shared__ int   i0[16][TOPK_], i1[16][TOPK_];
  const int t = threadIdx.x;
  const int wv = t >> 5, lane = t & 31;
  const int m = lane & 15, g = lane >> 4;
  const int l16 = lane & 15, g2 = lane >> 4;
  const int rb = blockIdx.x * 16;
  const int mrow = rb + m;                           // all 16 rows real
  const int myrow = wv * 2 + g2;                     // top-k: 16 lanes per row

  for (int d = 0; d < DEPTH_; ++d) {
    // A fragments (qn rows) held in registers across the whole N loop: 8 frags, K=256
    v16bf af[8];
    #pragma unroll
    for (int ks = 0; ks < 8; ++ks)
      af[ks] = load_frag(qn + (size_t)mrow * NQ_ + d * KDIM_ + ks * 32 + g * 8);

    for (int nt = wv; nt < 63; nt += 8) {            // 63 N-tiles cover 1008 >= 1000 keys
      int n = nt * 16 + m;
      int nc = n < NKEYS_ ? n : NKEYS_ - 1;
      const __bf16* kb = keysn + ((size_t)d * NKEYS_ + nc) * KDIM_ + g * 8;
      int n2 = (nt + 8) * 16 + m;                    // prefetch next tile's key row
      if (n2 < NKEYS_)
        __builtin_prefetch(keysn + ((size_t)d * NKEYS_ + n2) * KDIM_, 0, 0);
      // batch all 8 B fragments (load clause + single wait), then 8 WMMAs back-to-back
      v16bf bq[8];
      #pragma unroll
      for (int ks = 0; ks < 8; ++ks) bq[ks] = load_frag(kb + ks * 32);
      v8f acc = (v8f){0,0,0,0,0,0,0,0};
      #pragma unroll
      for (int ks = 0; ks < 8; ++ks)
        acc = __builtin_amdgcn_wmma_f32_16x16x32_bf16(
            false, af[ks], false, bq[ks], (short)0, acc, false, false);
      #pragma unroll
      for (int vr = 0; vr < 8; ++vr)
        logits[vr + 8 * g][nt * 16 + m] = acc[vr];
    }
    __syncthreads();
    {  // pad n in [1000,1024) with -inf: 24 slots per row
      int row = t >> 4, l = t & 15;
      logits[row][NKEYS_ + l] = NEGINF_;
      if (l < 8) logits[row][1016 + l] = NEGINF_;
    }
    __syncthreads();

    // iterative 16-lane argmax -> sorted top-32 for row myrow
    float* Lr = logits[myrow];
    float* S = d ? s1[myrow] : s0[myrow];
    int*   I = d ? i1[myrow] : i0[myrow];
    for (int iter = 0; iter < TOPK_; ++iter) {
      float bv = NEGINF_; int bn = 1023;
      for (int k2 = 0; k2 < 64; ++k2) {
        int pos = l16 + 16 * k2;
        float v = Lr[pos];
        if (v > bv) { bv = v; bn = pos; }
      }
      #pragma unroll
      for (int off = 8; off > 0; off >>= 1) {
        float ov = __shfl_xor(bv, off, 16);
        int   on = __shfl_xor(bn, off, 16);
        if (ov > bv || (ov == bv && on < bn)) { bv = ov; bn = on; }
      }
      if (l16 == 0) { S[iter] = bv; I[iter] = bn; Lr[bn] = NEGINF_; }
    }
    __syncthreads();
  }

  // cartesian combine: cand[i*32+j] = s0[i] + s1[j] (reuses logits LDS)
  {
    int row = t >> 4, ii = t & 15;
    #pragma unroll
    for (int h = 0; h < 2; ++h) {
      int i = ii + 16 * h;
      float si = s0[row][i];
      for (int j = 0; j < TOPK_; ++j)
        logits[row][i * TOPK_ + j] = si + s1[row][j];
    }
  }
  __syncthreads();

  // top-32 of 1024 candidates + softmax (selection order is descending)
  float vmax = 0.f, sumexp = 0.f;
  float myv[2] = {0.f, 0.f}; int myidx[2] = {0, 0};
  for (int iter = 0; iter < TOPK_; ++iter) {
    float bv = NEGINF_; int bn = 0;
    for (int k2 = 0; k2 < 64; ++k2) {
      int pos = l16 + 16 * k2;
      float v = logits[myrow][pos];
      if (v > bv) { bv = v; bn = pos; }
    }
    #pragma unroll
    for (int off = 8; off > 0; off >>= 1) {
      float ov = __shfl_xor(bv, off, 16);
      int   on = __shfl_xor(bn, off, 16);
      if (ov > bv || (ov == bv && on < bn)) { bv = ov; bn = on; }
    }
    if (iter == 0) vmax = bv;
    sumexp += expf(bv - vmax);
    if ((iter & 15) == l16) {
      int slot = iter >> 4;
      myv[slot] = bv;
      myidx[slot] = i0[myrow][bn >> 5] * NKEYS_ + i1[myrow][bn & 31];
    }
    if (l16 == 0) logits[myrow][bn] = NEGINF_;
  }
  int grow = rb + myrow;
  #pragma unroll
  for (int slot = 0; slot < 2; ++slot) {
    int e = slot * 16 + l16;
    out[(size_t)grow * TOPK_ + e] = expf(myv[slot] - vmax) / sumexp;
    out[(size_t)BS_ * TOPK_ + (size_t)grow * TOPK_ + e] = (float)myidx[slot];
  }
}

extern "C" void kernel_launch(void* const* d_in, const int* in_sizes, int n_in,
                              void* d_out, int out_size, void* d_ws, size_t ws_size,
                              hipStream_t stream) {
  (void)in_sizes; (void)n_in; (void)out_size; (void)ws_size;
  const float* x    = (const float*)d_in[0];
  const float* W    = (const float*)d_in[1];
  const float* b    = (const float*)d_in[2];
  const float* keys = (const float*)d_in[3];
  float* out = (float*)d_out;

  char* ws = (char*)d_ws;
  __bf16* Wbf   = (__bf16*)(ws);                          // 512*1024*2   = 1,048,576 B
  __bf16* keysn = (__bf16*)(ws + 1048576);                // 2*1000*256*2 = 1,024,000 B
  __bf16* qn    = (__bf16*)(ws + 1048576 + 1024000);      // 16384*512*2  = 16,777,216 B

  k_cvt_w      <<<512,      256, 0, stream>>>(W, Wbf);
  k_keys       <<<2000,     256, 0, stream>>>(keys, keysn);
  k_query      <<<BS_ / 32, 256, 0, stream>>>(x, Wbf, b, qn);
  k_logits_topk<<<BS_ / 16, 256, 0, stream>>>(qn, keysn, out);
}
